// MatrixLSTMCell_75265006895884
// MI455X (gfx1250) — matrix operand order repaired and checked by reading, not compiled, not run on hardware
//
#include <hip/hip_runtime.h>
#include <hip/hip_bf16.h>
#include <math.h>

// Problem constants (match reference)
#define B_    2
#define S_    2048
#define DIM_  1024
#define NH_   8
#define DH_   128
#define BH_   (B_ * NH_)
#define TQ_   128        // query rows per block (8 waves x 16 rows)
#define TK_   32         // keys per inner tile (K of the PV wmma)
#define SCALE_ 0.088388347648318447f   // 1/sqrt(128)
#define GCAP_  15.0f

typedef __attribute__((ext_vector_type(16))) __bf16 v16bf;
typedef __attribute__((ext_vector_type(8)))  __bf16 v8bf;
typedef __attribute__((ext_vector_type(8)))  float  v8f;

// Build a v16bf WMMA operand from two 16-byte-aligned 8-element LDS runs.
__device__ __forceinline__ v16bf ld_b16x16(const __bf16* p0, const __bf16* p1) {
  v8bf lo = *(const v8bf*)p0;
  v8bf hi = *(const v8bf*)p1;
  v16bf r;
#pragma unroll
  for (int j = 0; j < 8; ++j) { r[j] = lo[j]; r[8 + j] = hi[j]; }
  return r;
}

// ---------------------------------------------------------------------------
// Kernel 1: gate preactivations.  pre = [q,k,v] @ W^T + b, soft-capped.
// Writes i_pre (B,NH,S) and log_sigmoid(f_pre) (B,NH,S) to workspace.
// One 256-thread block per token.
// ---------------------------------------------------------------------------
__global__ __launch_bounds__(256) void gate_kernel(
    const float* __restrict__ q, const float* __restrict__ k,
    const float* __restrict__ v, const float* __restrict__ w,
    const float* __restrict__ bias,
    float* __restrict__ ipre, float* __restrict__ lfg) {
  const int token = blockIdx.x;
  const int b = token / S_, s = token % S_;
  const float* qr = q + (size_t)token * DIM_;
  const float* kr = k + (size_t)token * DIM_;
  const float* vr = v + (size_t)token * DIM_;

  float acc[16];
#pragma unroll
  for (int j = 0; j < 16; ++j) acc[j] = 0.f;

  for (int e = threadIdx.x; e < 3 * DIM_; e += 256) {
    const float x = (e < DIM_) ? qr[e] : (e < 2 * DIM_ ? kr[e - DIM_] : vr[e - 2 * DIM_]);
#pragma unroll
    for (int j = 0; j < 16; ++j) acc[j] += x * w[j * (3 * DIM_) + e];
  }
  // wave reduce (wave32)
#pragma unroll
  for (int m = 1; m < 32; m <<= 1) {
#pragma unroll
    for (int j = 0; j < 16; ++j) acc[j] += __shfl_xor(acc[j], m, 32);
  }
  __shared__ float red[8][16];
  const int wave = threadIdx.x >> 5, lane = threadIdx.x & 31;
  if (lane == 0) {
#pragma unroll
    for (int j = 0; j < 16; ++j) red[wave][j] = acc[j];
  }
  __syncthreads();
  if (threadIdx.x < 16) {
    const int j = threadIdx.x;
    float p = bias[j];
#pragma unroll
    for (int wv = 0; wv < 8; ++wv) p += red[wv][j];
    p = GCAP_ * tanhf(p * (1.f / GCAP_));
    if (j < 8) {
      ipre[((size_t)(b * NH_ + j)) * S_ + s] = p;             // input gate preact
    } else {
      const int h = j - 8;                                    // forget gate head
      const float ls = (p >= 0.f) ? -log1pf(__expf(-p)) : (p - log1pf(__expf(p)));
      lfg[((size_t)(b * NH_ + h)) * S_ + s] = ls;             // log_sigmoid(f)
    }
  }
}

// ---------------------------------------------------------------------------
// Kernel 2: per-(b,h) sequential scan.  C = inclusive cumsum(log_fg),
// g = i_pre - C, M = prefix-max(g).  One wave per (b,h).
// ---------------------------------------------------------------------------
__global__ __launch_bounds__(32) void scan_kernel(
    const float* __restrict__ ipre, const float* __restrict__ lfg,
    float* __restrict__ C, float* __restrict__ G, float* __restrict__ M) {
  const int bh = blockIdx.x, lane = threadIdx.x;
  const size_t base = (size_t)bh * S_;
  float carryC = 0.f, carryM = -INFINITY;
  for (int s0 = 0; s0 < S_; s0 += 32) {
    float x = lfg[base + s0 + lane];
#pragma unroll
    for (int d = 1; d < 32; d <<= 1) {
      const float y = __shfl_up(x, d, 32);
      if (lane >= d) x += y;
    }
    const float c = carryC + x;
    const float g = ipre[base + s0 + lane] - c;
    float mx = g;
#pragma unroll
    for (int d = 1; d < 32; d <<= 1) {
      const float y = __shfl_up(mx, d, 32);
      if (lane >= d) mx = fmaxf(mx, y);
    }
    mx = fmaxf(mx, carryM);
    C[base + s0 + lane] = c;
    G[base + s0 + lane] = g;
    M[base + s0 + lane] = mx;
    carryC = __shfl(c, 31, 32);
    carryM = __shfl(mx, 31, 32);
  }
}

// ---------------------------------------------------------------------------
// Kernel 3: fused mLSTM attention + denom + multi-head RMSNorm.
// grid (S/TQ, B*NH), 256 threads (8 waves); wave w owns query rows
// [t0+16w, t0+16w+16).  Both GEMMs use v_wmma_f32_16x16x32_bf16.
// ---------------------------------------------------------------------------
__global__ __launch_bounds__(256) void mlstm_kernel(
    const float* __restrict__ q, const float* __restrict__ k,
    const float* __restrict__ v,
    const float* __restrict__ C, const float* __restrict__ G,
    const float* __restrict__ M,
    const float* __restrict__ outw, float* __restrict__ out) {
  // K tile row-major (padded stride 136 bf16 = 272B, 16B aligned, bank-spread)
  __shared__ __bf16 kbuf[TK_][136];
  // V tile transposed: vtb[d][s] (stride 40 bf16 = 80B)
  __shared__ __bf16 vtb[DH_][40];
  // per-wave s_mat relayout scratch (D-layout -> A-layout bounce)
  __shared__ __bf16 sbuf[8][16][40];

  const int bh = blockIdx.y, b = bh / NH_, h = bh % NH_;
  const int t0 = blockIdx.x * TQ_;
  const int tid = threadIdx.x, wave = tid >> 5, lane = tid & 31;
  const int c = lane & 15, hf = lane >> 4;
  const int trow = t0 + wave * 16;
  const size_t gbase = (size_t)bh * S_;
  const float* qhead = q + (size_t)b * S_ * DIM_ + h * DH_;
  const float* khead = k + (size_t)b * S_ * DIM_ + h * DH_;
  const float* vhead = v + (size_t)b * S_ * DIM_ + h * DH_;

  // --- Q strip in WMMA A-layout (bf16), kept in registers all iterations.
  // A 16x32 bf16 layout: M = lane%16; half hf covers K = hf*8+[0,8) and 16+hf*8+[0,8).
  v16bf aq[4];
  {
    const float* qrow = qhead + (size_t)(trow + c) * DIM_;
#pragma unroll
    for (int kc = 0; kc < 4; ++kc) {
      const float* p = qrow + kc * 32 + hf * 8;
      const float4 a0 = *(const float4*)(p);
      const float4 a1 = *(const float4*)(p + 4);
      const float4 b0 = *(const float4*)(p + 16);
      const float4 b1 = *(const float4*)(p + 20);
      aq[kc][0]  = (__bf16)a0.x; aq[kc][1]  = (__bf16)a0.y;
      aq[kc][2]  = (__bf16)a0.z; aq[kc][3]  = (__bf16)a0.w;
      aq[kc][4]  = (__bf16)a1.x; aq[kc][5]  = (__bf16)a1.y;
      aq[kc][6]  = (__bf16)a1.z; aq[kc][7]  = (__bf16)a1.w;
      aq[kc][8]  = (__bf16)b0.x; aq[kc][9]  = (__bf16)b0.y;
      aq[kc][10] = (__bf16)b0.z; aq[kc][11] = (__bf16)b0.w;
      aq[kc][12] = (__bf16)b1.x; aq[kc][13] = (__bf16)b1.y;
      aq[kc][14] = (__bf16)b1.z; aq[kc][15] = (__bf16)b1.w;
    }
  }

  // Row-max stabilizers for this lane's 8 rows (D-layout rows = hf*8+i).
  float mrow[8];
#pragma unroll
  for (int i = 0; i < 8; ++i) mrow[i] = M[gbase + trow + hf * 8 + i];

  v8f acc[8];
#pragma unroll
  for (int n = 0; n < 8; ++n) {
#pragma unroll
    for (int i = 0; i < 8; ++i) acc[n][i] = 0.f;
  }
  float psum[8];
#pragma unroll
  for (int i = 0; i < 8; ++i) psum[i] = 0.f;

  const int sEnd = t0 + TQ_;   // keys s < sEnd (causal upper bound for this block)
  for (int s0 = 0; s0 < sEnd; s0 += TK_) {
    __syncthreads();
    {
      // Stage K (row-major bf16) and V (transposed bf16) into LDS.
      const int key = tid >> 3;          // 0..31
      const int dp  = (tid & 7) * 16;    // 0..112
      const float* krow = khead + (size_t)(s0 + key) * DIM_ + dp;
      const float* vrow = vhead + (size_t)(s0 + key) * DIM_ + dp;
#pragma unroll
      for (int j4 = 0; j4 < 4; ++j4) {
        const float4 kf = *(const float4*)(krow + j4 * 4);
        const float4 vf = *(const float4*)(vrow + j4 * 4);
        const int dd = dp + j4 * 4;
        kbuf[key][dd + 0] = (__bf16)kf.x; kbuf[key][dd + 1] = (__bf16)kf.y;
        kbuf[key][dd + 2] = (__bf16)kf.z; kbuf[key][dd + 3] = (__bf16)kf.w;
        vtb[dd + 0][key] = (__bf16)vf.x; vtb[dd + 1][key] = (__bf16)vf.y;
        vtb[dd + 2][key] = (__bf16)vf.z; vtb[dd + 3][key] = (__bf16)vf.w;
      }
    }
    __syncthreads();

    // ---- scores = Q * K^T for 16 rows x 32 keys (2 N-tiles x 4 K-chunks)
#pragma unroll
    for (int n2 = 0; n2 < 2; ++n2) {
      v8f sc;
#pragma unroll
      for (int i = 0; i < 8; ++i) sc[i] = 0.f;
#pragma unroll
      for (int kc = 0; kc < 4; ++kc) {
        const __bf16* kp = &kbuf[16 * n2 + c][kc * 32 + hf * 8];
        const v16bf bk = ld_b16x16(kp, kp + 16);
        sc = __builtin_amdgcn_wmma_f32_16x16x32_bf16(false, aq[kc], false, bk,
                                                     (short)0, sc, false, false);
      }
      // fp32 decay + causal mask + row-sum, then park bf16 s_mat in LDS.
      const int skey = s0 + 16 * n2 + c;
      const float gs = G[gbase + skey];
#pragma unroll
      for (int i = 0; i < 8; ++i) {
        const int t = trow + hf * 8 + i;
        const float d = (skey <= t) ? __expf(gs - mrow[i]) : 0.f;
        const float sd = sc[i] * SCALE_ * d;
        psum[i] += sd;
        sbuf[wave][hf * 8 + i][16 * n2 + c] = (__bf16)sd;
      }
    }
    // Per-wave LDS ops are in-order in HW; just stop the compiler reordering.
    asm volatile("" ::: "memory");

    // ---- re-layout s_mat into WMMA A-operand, then out += s_mat * V
    const v16bf sa = ld_b16x16(&sbuf[wave][c][hf * 8], &sbuf[wave][c][16 + hf * 8]);
#pragma unroll
    for (int n2 = 0; n2 < 8; ++n2) {
      const __bf16* vp = &vtb[16 * n2 + c][hf * 8];
      const v16bf bv = ld_b16x16(vp, vp + 16);
      acc[n2] = __builtin_amdgcn_wmma_f32_16x16x32_bf16(false, sa, false, bv,
                                                        (short)0, acc[n2], false, false);
    }
  }

  // ---- denominator: reduce row-sums across the 16 lanes of each half.
#pragma unroll
  for (int m = 1; m < 16; m <<= 1) {
#pragma unroll
    for (int i = 0; i < 8; ++i) psum[i] += __shfl_xor(psum[i], m, 32);
  }
  float rden[8];
#pragma unroll
  for (int i = 0; i < 8; ++i) {
    const int t = trow + hf * 8 + i;
    const float mm = C[gbase + t] + mrow[i];     // row max m[t] (|m| <= ~15)
    rden[i] = 1.f / (fmaxf(fabsf(psum[i]), __expf(-mm)) + 5e-5f);
  }

  // ---- h = num/denom, multi-head RMSNorm with residual gain (1 + w).
  float ssq[8];
#pragma unroll
  for (int i = 0; i < 8; ++i) ssq[i] = 0.f;
#pragma unroll
  for (int n2 = 0; n2 < 8; ++n2) {
#pragma unroll
    for (int i = 0; i < 8; ++i) {
      const float x = acc[n2][i] * rden[i];
      acc[n2][i] = x;
      ssq[i] += x * x;
    }
  }
#pragma unroll
  for (int m = 1; m < 16; m <<= 1) {
#pragma unroll
    for (int i = 0; i < 8; ++i) ssq[i] += __shfl_xor(ssq[i], m, 32);
  }
  float rr[8];
#pragma unroll
  for (int i = 0; i < 8; ++i) rr[i] = rsqrtf(ssq[i] * (1.f / DH_) + 1e-6f);

#pragma unroll
  for (int n2 = 0; n2 < 8; ++n2) {
    const float gw = 1.f + outw[h * DH_ + 16 * n2 + c];
#pragma unroll
    for (int i = 0; i < 8; ++i) {
      const int t = trow + hf * 8 + i;
      out[((size_t)(b * S_ + t)) * DIM_ + h * DH_ + 16 * n2 + c] =
          acc[n2][i] * rr[i] * gw;
    }
  }
}

// ---------------------------------------------------------------------------
extern "C" void kernel_launch(void* const* d_in, const int* in_sizes, int n_in,
                              void* d_out, int out_size, void* d_ws, size_t ws_size,
                              hipStream_t stream) {
  (void)in_sizes; (void)n_in; (void)out_size; (void)ws_size;
  const float* q     = (const float*)d_in[0];
  const float* k     = (const float*)d_in[1];
  const float* v     = (const float*)d_in[2];
  const float* gw    = (const float*)d_in[3];  // ifgate_w (16, 3072)
  const float* gb    = (const float*)d_in[4];  // ifgate_b (16,)
  const float* onw   = (const float*)d_in[5];  // outnorm_w (1024,)
  float*       out   = (float*)d_out;

  float* ws = (float*)d_ws;
  const size_t n = (size_t)BH_ * S_;           // 32768 floats per array
  float* ipre = ws;
  float* lfg  = ws + 1 * n;
  float* C    = ws + 2 * n;
  float* G    = ws + 3 * n;
  float* M    = ws + 4 * n;                    // total ws use: 640 KB

  gate_kernel<<<B_ * S_, 256, 0, stream>>>(q, k, v, gw, gb, ipre, lfg);
  scan_kernel<<<BH_, 32, 0, stream>>>(ipre, lfg, C, G, M);
  mlstm_kernel<<<dim3(S_ / TQ_, BH_), 256, 0, stream>>>(q, k, v, C, G, M, onw, out);
}
